// CrossAttention_59639915872332
// MI455X (gfx1250) — matrix operand order repaired and checked by reading, not compiled, not run on hardware
//
#include <hip/hip_runtime.h>
#include <hip/hip_bf16.h>

// ---------------------------------------------------------------------------
// CDNA5 (gfx1250) wave32 WMMA cross-attention pipeline.
//   Q = (x @ Wq) * scale ; K = ctx @ Wk ; V = ctx @ Wv   (bf16 WMMA -> bf16 ws)
//   O = softmax(Q Kt) V   (flash, fused)                 (bf16 WMMA -> bf16 ws)
//   out = O @ Wout + bout                                (bf16 WMMA -> fp32)
// ---------------------------------------------------------------------------

typedef __attribute__((ext_vector_type(16))) __bf16 v16bf;
typedef __attribute__((ext_vector_type(8)))  float  v8f;

#define ATTN_SCALE 0.125f  // 64^-0.5, folded into Q projection epilogue

__device__ __forceinline__ unsigned short f2bf(float f) {
    unsigned u = __float_as_uint(f);
    unsigned r = u + 0x7FFFu + ((u >> 16) & 1u);   // round-to-nearest-even
    return (unsigned short)(r >> 16);
}
__device__ __forceinline__ unsigned pack2bf(float a, float b) {
    return (unsigned)f2bf(a) | ((unsigned)f2bf(b) << 16);
}

// DPP ROW_XMASK butterfly step: lanes swap with (lane ^ XMASK) within rows of 16.
template <int XMASK>
__device__ __forceinline__ float row_xor(float x) {
    return __int_as_float(__builtin_amdgcn_update_dpp(
        0, __float_as_int(x), 0x160 | XMASK, 0xf, 0xf, true));
}
__device__ __forceinline__ float rowmax16(float x) {
    x = fmaxf(x, row_xor<1>(x));
    x = fmaxf(x, row_xor<2>(x));
    x = fmaxf(x, row_xor<4>(x));
    x = fmaxf(x, row_xor<8>(x));
    return x;
}
__device__ __forceinline__ float rowsum16(float x) {
    x += row_xor<1>(x);
    x += row_xor<2>(x);
    x += row_xor<4>(x);
    x += row_xor<8>(x);
    return x;
}

// 16 contiguous bf16 (32B) from a 16B-aligned LDS address.
__device__ __forceinline__ v16bf frag16(const unsigned short* p) {
    union { struct { uint4 a, b; } u; v16bf v; } f;
    f.u.a = *(const uint4*)(p);
    f.u.b = *(const uint4*)(p + 8);
    return f.v;
}
// Two 16B halves with a gap (A fragments pass gap 8: K = base..base+15 contiguous, base = 16*half).
__device__ __forceinline__ v16bf frag_split(const unsigned short* p, int gap) {
    union { struct { uint4 a, b; } u; v16bf v; } f;
    f.u.a = *(const uint4*)(p);
    f.u.b = *(const uint4*)(p + gap);
    return f.v;
}

__device__ __forceinline__ v8f wmma_bf16(v16bf a, v16bf b, v8f c) {
    // D = A(16x32) * B(32x16) + C, fp32 accumulate
    return __builtin_amdgcn_wmma_f32_16x16x32_bf16(
        /*neg_a=*/false, a, /*neg_b=*/false, b,
        /*c_mod=*/(short)0, c, /*reuse_a=*/false, /*reuse_b=*/false);
}

// ---------------------------------------------------------------------------
// Generic tiled GEMM:  Out[M,N] = (A[M,K] * B[K,N]) * oscale (+bias)
// Block tile 128(M) x 64(N), K-step 32, 256 threads = 8 waves, wave does 16x64.
// ---------------------------------------------------------------------------
#define LDA 40   // 32 + 8 pad (ushorts); 80B row stride, 16B aligned
#define LDB 40

template <bool A_BF16, bool OUT_BF16, bool BIAS>
__global__ __launch_bounds__(256) void gemm_wmma(
    const void* __restrict__ Ap, const float* __restrict__ B,
    const float* __restrict__ bias, void* __restrict__ Outp,
    int M, int K, int N, float oscale) {
    __shared__ unsigned short As[128 * LDA];
    __shared__ unsigned short Bs[64 * LDB];

    const int tid  = threadIdx.x;
    const int lane = tid & 31;
    const int wave = tid >> 5;        // 0..7 -> M sub-tile
    const int half = lane >> 4;       // 0/1
    const int l16  = lane & 15;
    const int ko   = half ? 16 : 0;   // A-frag K base
    const int koff = half ? 16 : 0;   // B-frag K base

    const int mblk = blockIdx.y * 128;
    const int nblk = blockIdx.x * 64;

    const float*          Af = (const float*)Ap;
    const unsigned short* Ab = (const unsigned short*)Ap;

    v8f acc[4] = {};

    const int atr = tid >> 3;          // A loader: 32 rows/pass, 8 thr x4 elems per row
    const int atc = (tid & 7) << 2;
    const int btr = tid >> 4;          // B loader: 16 k-rows/pass, 16 thr x4 f32 per row
    const int btc = (tid & 15) << 2;

    for (int k0 = 0; k0 < K; k0 += 32) {
        // ---- stage A tile 128x32 (fp32 converted, or bf16 passed through) ----
#pragma unroll
        for (int p = 0; p < 4; ++p) {
            int row = p * 32 + atr;
            long gi = (long)(mblk + row) * K + k0 + atc;
            uint2 w;
            if (A_BF16) {
                w = *(const uint2*)(Ab + gi);        // already bf16 pairs
            } else {
                float4 f = *(const float4*)(Af + gi);
                w.x = pack2bf(f.x, f.y);
                w.y = pack2bf(f.z, f.w);
            }
            *(uint2*)&As[row * LDA + atc] = w;       // single ds_store_b64
        }
        // ---- stage B tile 32x64, transposed -> Bs[n][k] ----
#pragma unroll
        for (int p = 0; p < 2; ++p) {
            int kr = p * 16 + btr;
            float4 f = *(const float4*)(B + (long)(k0 + kr) * N + nblk + btc);
            Bs[(btc + 0) * LDB + kr] = f2bf(f.x);
            Bs[(btc + 1) * LDB + kr] = f2bf(f.y);
            Bs[(btc + 2) * LDB + kr] = f2bf(f.z);
            Bs[(btc + 3) * LDB + kr] = f2bf(f.w);
        }
        __syncthreads();

        v16bf a = frag_split(&As[(wave * 16 + l16) * LDA + ko], 8);
#pragma unroll
        for (int nt = 0; nt < 4; ++nt) {
            v16bf b = frag16(&Bs[(nt * 16 + l16) * LDB + koff]);
            acc[nt] = wmma_bf16(a, b, acc[nt]);
        }
        __syncthreads();
    }

    // ---- epilogue: C layout VGPR r -> M = r + 8*half, N = l16 ----
    const int orow0 = mblk + wave * 16 + half * 8;
#pragma unroll
    for (int nt = 0; nt < 4; ++nt) {
        int col  = nblk + nt * 16 + l16;
        float bv = BIAS ? bias[col] : 0.0f;
#pragma unroll
        for (int r = 0; r < 8; ++r) {
            float v  = acc[nt][r] * oscale + bv;
            long idx = (long)(orow0 + r) * N + col;
            if (OUT_BF16) ((unsigned short*)Outp)[idx] = f2bf(v);
            else          ((float*)Outp)[idx]          = v;
        }
    }
}

// ---------------------------------------------------------------------------
// Fused flash attention per (b, h, 64-query tile). 128 threads = 4 waves,
// wave owns 16 query rows x full d=64. m=1024 keys in 32-key blocks.
// Q/K/V/O are bf16 workspace, row-major [rows][512], head slice = cols h*64..+63.
// Q already carries the 1/sqrt(d) scale.
// ---------------------------------------------------------------------------
__global__ __launch_bounds__(128) void attn_flash(
    const unsigned short* __restrict__ Q, const unsigned short* __restrict__ Kw,
    const unsigned short* __restrict__ Vw, unsigned short* __restrict__ O) {
    const int qt = blockIdx.x;   // 0..31 query tile
    const int h  = blockIdx.y;   // 0..7
    const int b  = blockIdx.z;   // 0..3

    __shared__ unsigned short Qs[64 * 72];       // [q][d]   144B rows
    __shared__ unsigned short Ks[32 * 72];       // [key][d] (natural = QK^T B-layout)
    __shared__ unsigned short Vt[64 * 40];       // [d][key] (PV B-layout)
    __shared__ unsigned short Ps[4][16 * 40];    // per-wave P 16x32 staging

    const int tid  = threadIdx.x;
    const int lane = tid & 31;
    const int wave = tid >> 5;       // 0..3 -> query rows wave*16..+15
    const int half = lane >> 4;
    const int l16  = lane & 15;
    const int ko   = half ? 16 : 0;
    const int koff = half ? 16 : 0;

    const int  hcol  = h * 64;
    const long qrow0 = (long)b * 2048 + (long)qt * 64;
    const long krow0 = (long)b * 1024;

    const int lr = tid >> 4;          // loader: 8 rows/pass
    const int lc = (tid & 15) << 2;   // 4 ushorts per thread per row

    // ---- stage Q tile 64x64 once ----
#pragma unroll
    for (int p = 0; p < 8; ++p) {
        int row = p * 8 + lr;
        uint2 u = *(const uint2*)(Q + (qrow0 + row) * 512 + hcol + lc);
        *(uint2*)&Qs[row * 72 + lc] = u;
    }
    __syncthreads();

    // Q fragments live in registers for the whole key loop (2 d-chunks of 32).
    v16bf qa0 = frag_split(&Qs[(wave * 16 + l16) * 72 + 0 * 32 + ko], 8);
    v16bf qa1 = frag_split(&Qs[(wave * 16 + l16) * 72 + 1 * 32 + ko], 8);

    v8f   o[4] = {};
    float mrow[8], lrow[8];
#pragma unroll
    for (int r = 0; r < 8; ++r) { mrow[r] = -__builtin_inff(); lrow[r] = 0.0f; }

    for (int kb = 0; kb < 32; ++kb) {
        __syncthreads();   // prior iteration's WMMA reads of Ks/Vt are done
        // ---- stage K tile 32x64 and V tile transposed ----
#pragma unroll
        for (int p = 0; p < 4; ++p) {
            int key = p * 8 + lr;
            long grow = krow0 + kb * 32 + key;
            uint2 u = *(const uint2*)(Kw + grow * 512 + hcol + lc);
            *(uint2*)&Ks[key * 72 + lc] = u;
            uint2 v = *(const uint2*)(Vw + grow * 512 + hcol + lc);
            Vt[(lc + 0) * 40 + key] = (unsigned short)(v.x & 0xffffu);
            Vt[(lc + 1) * 40 + key] = (unsigned short)(v.x >> 16);
            Vt[(lc + 2) * 40 + key] = (unsigned short)(v.y & 0xffffu);
            Vt[(lc + 3) * 40 + key] = (unsigned short)(v.y >> 16);
        }
        // prefetch next key block while this one computes (global_prefetch_b8)
        if (kb + 1 < 32) {
            long nrow = krow0 + (kb + 1) * 32 + lr;
            __builtin_prefetch(Kw + nrow * 512 + hcol + lc, 0, 0);
            __builtin_prefetch(Vw + nrow * 512 + hcol + lc, 0, 0);
        }
        __syncthreads();

        // ---- S = Q * K^T : s0 = keys 0..15, s1 = keys 16..31 ----
        v8f s0 = {}, s1 = {};
        {
            v16bf k0 = frag16(&Ks[(0 * 16 + l16) * 72 + 0 * 32 + koff]);
            v16bf k1 = frag16(&Ks[(1 * 16 + l16) * 72 + 0 * 32 + koff]);
            s0 = wmma_bf16(qa0, k0, s0);
            s1 = wmma_bf16(qa0, k1, s1);
            k0 = frag16(&Ks[(0 * 16 + l16) * 72 + 1 * 32 + koff]);
            k1 = frag16(&Ks[(1 * 16 + l16) * 72 + 1 * 32 + koff]);
            s0 = wmma_bf16(qa1, k0, s0);
            s1 = wmma_bf16(qa1, k1, s1);
        }

        // ---- online softmax (row = r + 8*half, spread over 16 lanes) ----
        // DPP ROW_XMASK butterflies: pure VALU, no LDS traffic.
#pragma unroll
        for (int r = 0; r < 8; ++r) {
            float a0 = s0[r];
            float a1 = s1[r];
            float mx  = rowmax16(fmaxf(a0, a1));
            float nm  = fmaxf(mrow[r], mx);
            float fac = __expf(mrow[r] - nm);
            mrow[r] = nm;
            float p0 = __expf(a0 - nm);
            float p1 = __expf(a1 - nm);
            float sm = rowsum16(p0 + p1);
            lrow[r] = lrow[r] * fac + sm;
            o[0][r] *= fac; o[1][r] *= fac; o[2][r] *= fac; o[3][r] *= fac;
            // stage P (bf16) for the PV A-fragment re-layout (wave-private)
            unsigned short* pd = &Ps[wave][(r + half * 8) * 40];
            pd[l16 + 0]  = f2bf(p0);
            pd[l16 + 16] = f2bf(p1);
        }

        // ---- O += P(16x32) * V(32x64) ---- (same-wave LDS RAW: in-order DS)
        v16bf pa = frag_split(&Ps[wave][l16 * 40 + ko], 8);
#pragma unroll
        for (int nt = 0; nt < 4; ++nt) {
            v16bf vb = frag16(&Vt[(nt * 16 + l16) * 40 + koff]);
            o[nt] = wmma_bf16(pa, vb, o[nt]);
        }
    }

    // ---- normalize and write bf16 attention output ----
#pragma unroll
    for (int r = 0; r < 8; ++r) {
        float inv = 1.0f / lrow[r];
        long  row = qrow0 + wave * 16 + half * 8 + r;
#pragma unroll
        for (int nt = 0; nt < 4; ++nt) {
            O[row * 512 + hcol + nt * 16 + l16] = f2bf(o[nt][r] * inv);
        }
    }
}

// ---------------------------------------------------------------------------
extern "C" void kernel_launch(void* const* d_in, const int* in_sizes, int n_in,
                              void* d_out, int out_size, void* d_ws, size_t ws_size,
                              hipStream_t stream) {
    (void)in_sizes; (void)n_in; (void)out_size; (void)ws_size;
    const float* x    = (const float*)d_in[0];  // [4,2048,1024]
    const float* ctx  = (const float*)d_in[1];  // [4,1024,768]
    const float* Wq   = (const float*)d_in[2];  // [1024,512]
    const float* Wk   = (const float*)d_in[3];  // [768,512]
    const float* Wv   = (const float*)d_in[4];  // [768,512]
    const float* Wout = (const float*)d_in[5];  // [512,1024]
    const float* bout = (const float*)d_in[6];  // [1024]
    float* out = (float*)d_out;                 // [4,2048,1024]

    unsigned short* Qws = (unsigned short*)d_ws;              // 8192x512 bf16
    unsigned short* Kws = Qws + (size_t)8192 * 512;           // 4096x512
    unsigned short* Vws = Kws + (size_t)4096 * 512;           // 4096x512
    unsigned short* Ows = Vws + (size_t)4096 * 512;           // 8192x512

    dim3 blk(256);
    // Q = (x @ Wq) * ATTN_SCALE   (M=8192, K=1024, N=512)
    gemm_wmma<false, true, false><<<dim3(512 / 64, 8192 / 128), blk, 0, stream>>>(
        x, Wq, nullptr, Qws, 8192, 1024, 512, ATTN_SCALE);
    // K = ctx @ Wk    (M=4096, K=768, N=512)
    gemm_wmma<false, true, false><<<dim3(512 / 64, 4096 / 128), blk, 0, stream>>>(
        ctx, Wk, nullptr, Kws, 4096, 768, 512, 1.0f);
    // V = ctx @ Wv
    gemm_wmma<false, true, false><<<dim3(512 / 64, 4096 / 128), blk, 0, stream>>>(
        ctx, Wv, nullptr, Vws, 4096, 768, 512, 1.0f);
    // fused flash attention: grid (qtiles=32, heads=8, batch=4)
    attn_flash<<<dim3(32, 8, 4), dim3(128), 0, stream>>>(Qws, Kws, Vws, Ows);
    // out = O @ Wout + bout   (M=8192, K=512, N=1024)
    gemm_wmma<true, false, true><<<dim3(1024 / 64, 8192 / 128), blk, 0, stream>>>(
        Ows, Wout, bout, out, 8192, 512, 1024, 1.0f);
}
